// H0_78683800862947
// MI455X (gfx1250) — compile-verified, matrix operand order checked by reading, not run on hardware
//
#include <hip/hip_runtime.h>
#include <hip/hip_bf16.h>
#include <math.h>

#define Tn 4096
#define Bn 32
#define Fn 3
#define Ln 4
#define EPSf 1e-6f
#define ROWS (Tn*Bn)   /* 131072 */

typedef __attribute__((ext_vector_type(16))) _Float16 v16h;
typedef __attribute__((ext_vector_type(8)))  _Float16 v8h;
typedef __attribute__((ext_vector_type(8)))  float    v8f;
typedef __attribute__((ext_vector_type(4)))  unsigned int v4u;
typedef __attribute__((ext_vector_type(8)))  int      v8i;
typedef __attribute__((ext_vector_type(4)))  int      v4i;

#if defined(__has_builtin)
#if __has_builtin(__builtin_amdgcn_tensor_load_to_lds) && \
    __has_builtin(__builtin_amdgcn_s_wait_tensorcnt)
#define HAS_TDM 1
#endif
#endif
#ifndef HAS_TDM
#define HAS_TDM 0
#endif

// ---- WMMA helper (CDNA5 16x16x32 f16 -> f32) --------------------------------
__device__ __forceinline__ v8f wmma16(v16h a, v16h b, v8f c) {
  return __builtin_amdgcn_wmma_f32_16x16x32_f16(false, a, false, b, (short)0, c,
                                                false, false);
}

// A fragment (16x32 f16) gathered from a row-major tile (LDS or global).
// Lanes 0-15 hold K 0-7 / 16-23; lanes 16-31 hold K 8-15 / 24-31.
__device__ __forceinline__ v16h load_a_frag(const _Float16* A, int lda,
                                            int m0, int k0, int lane) {
  const _Float16* p = A + (size_t)(m0 + (lane & 15)) * lda
                        + (k0 + ((lane >> 4) << 3));
  v8h lo = *(const v8h*)p;
  v8h hi = *(const v8h*)(p + 16);
  v16h f;
#pragma unroll
  for (int i = 0; i < 8; ++i) { f[i] = lo[i]; f[i + 8] = hi[i]; }
  return f;
}

__device__ __forceinline__ unsigned lds_off_u32(const void* p) {
  // LDS flat addresses map through the low 32 bits (ISA 10.2 aperture rules).
  return (unsigned)(unsigned long long)p;
}

#if HAS_TDM
// Issue one 2-D TDM load: rows x kElems f16 tile, row stride = kElems,
// global -> LDS.  D# bitfields per CDNA5 ISA.
__device__ __forceinline__ void tdm_load_a_f16(const _Float16* gsrc,
                                               unsigned ldsByte,
                                               unsigned kElems, unsigned rows) {
  unsigned long long ga = (unsigned long long)gsrc;
  v4u g0;
  g0[0] = 1u;                                             // count=1, user mode
  g0[1] = ldsByte;                                        // lds_addr
  g0[2] = (unsigned)(ga & 0xFFFFFFFFull);                 // global_addr lo
  g0[3] = (unsigned)((ga >> 32) & 0x01FFFFFFull)          // global_addr hi
        | 0x80000000u;                                    // type=2 (image)
  v8i g1;
  g1[0] = (int)(1u << 16);                                // data_size=1 (2B)
  g1[1] = (int)((kElems & 0xFFFFu) << 16);                // tensor_dim0 lo16
  g1[2] = (int)((kElems >> 16) & 0xFFFFu)                 // tensor_dim0 hi16
        | (int)((rows & 0xFFFFu) << 16);                  // tensor_dim1 lo16
  g1[3] = (int)((rows >> 16) & 0xFFFFu)                   // tensor_dim1 hi16
        | (int)((kElems & 0xFFFFu) << 16);                // tile_dim0
  g1[4] = (int)(rows & 0xFFFFu);                          // tile_dim1 (dim2=0)
  g1[5] = (int)kElems;                                    // dim0_stride lo32
  g1[6] = 0;                                              // stride hi / dim1_stride lo
  g1[7] = 0;
  v4i z4 = {0, 0, 0, 0};
#if __clang_major__ >= 23
  v8i z8 = {0, 0, 0, 0, 0, 0, 0, 0};
  __builtin_amdgcn_tensor_load_to_lds(g0, g1, z4, z4, z8, 0);
#else
  __builtin_amdgcn_tensor_load_to_lds(g0, g1, z4, z4, 0);
#endif
}
#endif

// Stage a rows x kElems f16 tile into LDS (TDM if available, else cooperative).
__device__ __forceinline__ void stage_a_tile(_Float16* lds, const _Float16* g,
                                             int kElems, int rows, int nthreads) {
#if HAS_TDM
  if (threadIdx.x < 32) tdm_load_a_f16(g, lds_off_u32(lds), kElems, rows);
  __builtin_amdgcn_s_wait_tensorcnt(0);
#else
  v8h* dv = (v8h*)lds;
  const v8h* sv = (const v8h*)g;
  int tot = rows * kElems / 8;
  for (int i = threadIdx.x; i < tot; i += nthreads) dv[i] = sv[i];
#endif
  __syncthreads();
}

// ---- Weight pre-pack: f32 (K x N row-major) -> f16 B-fragment order ---------
__global__ void k_pack_b(const float* __restrict__ W, int K, int N, int Npad,
                         _Float16* __restrict__ dst) {
  int idx = blockIdx.x;          // fragment index = nb*KB + kb
  int KB = K >> 5;
  int kb = idx % KB, nb = idx / KB;
  int lane = threadIdx.x;        // 32 threads
  int n = nb * 16 + (lane & 15);
  _Float16* out = dst + ((size_t)idx * 32 + lane) * 16;
#pragma unroll
  for (int hi = 0; hi < 16; ++hi) {
    int kl = (hi & 7) + ((hi >> 3) << 4) + ((lane >> 4) << 3);
    int k = kb * 32 + kl;
    float v = (n < N) ? W[(size_t)k * N + n] : 0.0f;
    out[hi] = (_Float16)v;
  }
}

// ---- 1. EMA-standardize scan (serial over T, parallel over B*F*L=384) -------
__global__ void k_ema(const float* __restrict__ x, float* __restrict__ xe) {
  int c = threadIdx.x;
  if (c >= Bn * Fn * Ln) return;
  int b = c / (Fn * Ln);
  int fl = c % (Fn * Ln);
  int f = fl / Ln, l = fl % Ln;
  float a = exp2f(-(float)(l + 1));
  float m = 0.f, v = 0.f;
  for (int t = 0; t < Tn; ++t) {
    float xt = x[((size_t)t * Bn + b) * Fn + f];
    m = a * xt + (1.f - a) * m;
    v = a * xt * xt + (1.f - a) * v;
    float var = fmaxf(v - m * m, 0.f);
    xe[((size_t)t * Bn + b) * (Fn * Ln) + fl] = (xt - m) * rsqrtf(var + EPSf);
  }
}

// ---- 2. h = xe@W_in + b_in (K=12, VALU), fused RMSNorm -> hn (f16) ----------
__global__ void k_in_norm(const float* __restrict__ xe,
                          const float* __restrict__ W_in,
                          const float* __restrict__ b_in,
                          const float* __restrict__ scale,
                          float* __restrict__ h, _Float16* __restrict__ hn16) {
  int lane = threadIdx.x & 31;
  int wv = threadIdx.x >> 5;
  size_t row = (size_t)blockIdx.x * 8 + wv;   // one wave per row
  const float* xr = xe + row * 12;
  float a0 = b_in[lane], a1 = b_in[lane + 32];
#pragma unroll
  for (int j = 0; j < 12; ++j) {
    float xv = xr[j];
    a0 = fmaf(xv, W_in[j * 64 + lane], a0);
    a1 = fmaf(xv, W_in[j * 64 + lane + 32], a1);
  }
  float ss = a0 * a0 + a1 * a1;
#pragma unroll
  for (int o = 16; o > 0; o >>= 1) ss += __shfl_xor(ss, o, 32);
  float r = rsqrtf(ss * (1.0f / 64.0f) + EPSf);
  h[row * 64 + lane] = a0;
  h[row * 64 + lane + 32] = a1;
  hn16[row * 64 + lane] = (_Float16)(a0 * r * scale[lane]);
  hn16[row * 64 + lane + 32] = (_Float16)(a1 * r * scale[lane + 32]);
}

// ---- 3. u = hn @ mW_in + mb_in  (K=64, N=256) -- TDM stage + WMMA -----------
// Block: 8 waves, BM=32 rows.  Wave w: m-tile w>>2, n-tiles (w&3)*4 .. +3.
__global__ void k_gemm_u(const _Float16* __restrict__ A,
                         const _Float16* __restrict__ Bp,
                         const float* __restrict__ mb_in,
                         _Float16* __restrict__ u16) {
  __shared__ _Float16 As[32 * 64];
  int lane = threadIdx.x & 31, wv = threadIdx.x >> 5;
  size_t mblk = (size_t)blockIdx.x * 32;
  stage_a_tile(As, A + mblk * 64, 64, 32, 256);
  int mrel = (wv >> 2) * 16;
  v16h af0 = load_a_frag(As, 64, mrel, 0, lane);
  v16h af1 = load_a_frag(As, 64, mrel, 32, lane);
  const v16h* B = (const v16h*)Bp;
  int nbase = (wv & 3) * 4;
#pragma unroll
  for (int q = 0; q < 4; ++q) {
    int nb = nbase + q;
    v8f acc = {};
    acc = wmma16(af0, B[(size_t)(nb * 2 + 0) * 32 + lane], acc);
    acc = wmma16(af1, B[(size_t)(nb * 2 + 1) * 32 + lane], acc);
    int n = nb * 16 + (lane & 15);
    float bias = mb_in[n];
#pragma unroll
    for (int v = 0; v < 8; ++v) {
      size_t m = mblk + mrel + v + ((lane >> 4) << 3);
      u16[m * 256 + n] = (_Float16)(acc[v] + bias);
    }
  }
}

// ---- 4. causal depthwise conv (KC=4) + SiLU -> xc (f16) ---------------------
__global__ void k_conv(const _Float16* __restrict__ u16,
                       const float* __restrict__ convW,
                       const float* __restrict__ convb,
                       _Float16* __restrict__ xc16) {
  size_t idx = (size_t)blockIdx.x * blockDim.x + threadIdx.x;
  int d = (int)(idx & 127);
  size_t tb = idx >> 7;
  int b = (int)(tb % Bn);
  int t = (int)(tb / Bn);
  float acc = convb[d];
#pragma unroll
  for (int k = 0; k < 4; ++k) {
    int tt = t - 3 + k;
    if (tt >= 0)
      acc = fmaf(convW[d * 4 + k], (float)u16[((size_t)tt * Bn + b) * 256 + d], acc);
  }
  acc = acc / (1.f + expf(-acc));   // silu
  xc16[((size_t)t * Bn + b) * 128 + d] = (_Float16)acc;
}

// ---- 5. xp = xc @ W_xp  (K=128, N=36 pad 48) -- TDM stage + WMMA ------------
// Block: 8 waves, BM=128 rows.  Wave w: m-tile w, n-tiles 0..2.
__global__ void k_gemm_xp(const _Float16* __restrict__ A,
                          const _Float16* __restrict__ Bp,
                          float* __restrict__ xp) {
  __shared__ _Float16 As[128 * 128];
  int lane = threadIdx.x & 31, wv = threadIdx.x >> 5;
  size_t mblk = (size_t)blockIdx.x * 128;
  stage_a_tile(As, A + mblk * 128, 128, 128, 256);
  int mrel = wv * 16;
  v16h af[4];
#pragma unroll
  for (int kb = 0; kb < 4; ++kb)
    af[kb] = load_a_frag(As, 128, mrel, kb * 32, lane);
  const v16h* B = (const v16h*)Bp;
#pragma unroll
  for (int nb = 0; nb < 3; ++nb) {
    v8f acc = {};
#pragma unroll
    for (int kb = 0; kb < 4; ++kb)
      acc = wmma16(af[kb], B[(size_t)(nb * 4 + kb) * 32 + lane], acc);
    int n = nb * 16 + (lane & 15);
#pragma unroll
    for (int v = 0; v < 8; ++v) {
      size_t m = mblk + mrel + v + ((lane >> 4) << 3);
      xp[m * 48 + n] = acc[v];
    }
  }
}

// ---- 6. selective scan + delta(softplus) + gating, 4096 parallel chans ------
__global__ void k_scan(const float* __restrict__ xp,
                       const _Float16* __restrict__ xc16,
                       const _Float16* __restrict__ u16,
                       const float* __restrict__ W_dt,
                       const float* __restrict__ b_dt,
                       const float* __restrict__ A_log,
                       const float* __restrict__ Dp,
                       _Float16* __restrict__ yg16) {
  int b = blockIdx.x;       // 32 blocks
  int d = threadIdx.x;      // 128 threads
  float w0 = W_dt[0 * 128 + d], w1 = W_dt[1 * 128 + d];
  float w2 = W_dt[2 * 128 + d], w3 = W_dt[3 * 128 + d];
  float bdt = b_dt[d], dp = Dp[d];
  float Arow[16], h[16];
#pragma unroll
  for (int s = 0; s < 16; ++s) { Arow[s] = -expf(A_log[d * 16 + s]); h[s] = 0.f; }
  for (int t = 0; t < Tn; ++t) {
    size_t base = (size_t)t * Bn + b;
    const float* xr = xp + base * 48;
    float dl = bdt + w0 * xr[0] + w1 * xr[1] + w2 * xr[2] + w3 * xr[3];
    float delta = (dl > 20.f) ? dl : log1pf(expf(dl));  // softplus
    float xt = (float)xc16[base * 128 + d];
    float dx = delta * xt;
    float y = 0.f;
#pragma unroll
    for (int s = 0; s < 16; ++s) {
      h[s] = expf(delta * Arow[s]) * h[s] + dx * xr[4 + s];   // B_t
      y = fmaf(h[s], xr[20 + s], y);                          // C_t
    }
    y = fmaf(dp, xt, y);
    float res = (float)u16[base * 256 + 128 + d];
    y *= res / (1.f + expf(-res));                            // * silu(res)
    yg16[base * 128 + d] = (_Float16)y;
  }
}

// ---- 7. hfin = h + yg @ W_out + b_out  (K=128, N=64) -- TDM stage + WMMA ----
// Block: 8 waves, BM=64 rows.  Wave w: m-tile w>>1, n-tiles (w&1)*2 .. +1.
__global__ void k_gemm_out(const _Float16* __restrict__ A,
                           const _Float16* __restrict__ Bp,
                           const float* __restrict__ b_out,
                           const float* __restrict__ hres,
                           float* __restrict__ hfin) {
  __shared__ _Float16 As[64 * 128];
  int lane = threadIdx.x & 31, wv = threadIdx.x >> 5;
  size_t mblk = (size_t)blockIdx.x * 64;
  stage_a_tile(As, A + mblk * 128, 128, 64, 256);
  int mrel = (wv >> 1) * 16;
  v16h af[4];
#pragma unroll
  for (int kb = 0; kb < 4; ++kb)
    af[kb] = load_a_frag(As, 128, mrel, kb * 32, lane);
  const v16h* B = (const v16h*)Bp;
  int nbase = (wv & 1) * 2;
#pragma unroll
  for (int q = 0; q < 2; ++q) {
    int nb = nbase + q;
    v8f acc = {};
#pragma unroll
    for (int kb = 0; kb < 4; ++kb)
      acc = wmma16(af[kb], B[(size_t)(nb * 4 + kb) * 32 + lane], acc);
    int n = nb * 16 + (lane & 15);
    float bias = b_out[n];
#pragma unroll
    for (int v = 0; v < 8; ++v) {
      size_t m = mblk + mrel + v + ((lane >> 4) << 3);
      hfin[m * 64 + n] = acc[v] + bias + hres[m * 64 + n];
    }
  }
}

// ---- 8. out = hfin @ W_sim + b_sim  (N=3, VALU) -----------------------------
__global__ void k_final(const float* __restrict__ hfin,
                        const float* __restrict__ W_sim,
                        const float* __restrict__ b_sim,
                        float* __restrict__ out) {
  size_t row = (size_t)blockIdx.x * blockDim.x + threadIdx.x;
  const float* hr = hfin + row * 64;
  float a0 = b_sim[0], a1 = b_sim[1], a2 = b_sim[2];
#pragma unroll
  for (int dd = 0; dd < 64; ++dd) {
    float hv = hr[dd];
    a0 = fmaf(hv, W_sim[dd * 3 + 0], a0);
    a1 = fmaf(hv, W_sim[dd * 3 + 1], a1);
    a2 = fmaf(hv, W_sim[dd * 3 + 2], a2);
  }
  out[row * 3 + 0] = a0;
  out[row * 3 + 1] = a1;
  out[row * 3 + 2] = a2;
}

// ---- launcher ---------------------------------------------------------------
extern "C" void kernel_launch(void* const* d_in, const int* in_sizes, int n_in,
                              void* d_out, int out_size, void* d_ws, size_t ws_size,
                              hipStream_t stream) {
  (void)in_sizes; (void)n_in; (void)out_size; (void)ws_size;
  const float* x     = (const float*)d_in[0];
  const float* W_in  = (const float*)d_in[1];
  const float* b_in  = (const float*)d_in[2];
  const float* rms_s = (const float*)d_in[3];
  const float* mW_in = (const float*)d_in[4];
  const float* mb_in = (const float*)d_in[5];
  const float* convW = (const float*)d_in[6];
  const float* convb = (const float*)d_in[7];
  const float* W_xp  = (const float*)d_in[8];
  const float* W_dt  = (const float*)d_in[9];
  const float* b_dt  = (const float*)d_in[10];
  const float* A_log = (const float*)d_in[11];
  const float* Dp    = (const float*)d_in[12];
  const float* W_out = (const float*)d_in[13];
  const float* b_out = (const float*)d_in[14];
  const float* W_sim = (const float*)d_in[15];
  const float* b_sim = (const float*)d_in[16];
  float* out = (float*)d_out;

  char* ws = (char*)d_ws;
  size_t off = 0;
  auto alloc = [&](size_t bytes) -> void* {
    void* p = ws + off;
    off += (bytes + 255) & ~(size_t)255;
    return p;
  };
  float*    xe   = (float*)alloc((size_t)ROWS * 12 * 4);
  float*    h    = (float*)alloc((size_t)ROWS * 64 * 4);
  _Float16* hn16 = (_Float16*)alloc((size_t)ROWS * 64 * 2);
  _Float16* u16  = (_Float16*)alloc((size_t)ROWS * 256 * 2);
  _Float16* xc16 = (_Float16*)alloc((size_t)ROWS * 128 * 2);
  float*    xpb  = (float*)alloc((size_t)ROWS * 48 * 4);
  _Float16* yg16 = (_Float16*)alloc((size_t)ROWS * 128 * 2);
  float*    hfin = (float*)alloc((size_t)ROWS * 64 * 4);
  _Float16* pB2  = (_Float16*)alloc((size_t)32 * 32 * 16 * 2);  // K=64,N=256
  _Float16* pB3  = (_Float16*)alloc((size_t)12 * 32 * 16 * 2);  // K=128,Npad=48
  _Float16* pB5  = (_Float16*)alloc((size_t)16 * 32 * 16 * 2);  // K=128,N=64

  k_pack_b<<<32, 32, 0, stream>>>(mW_in, 64, 256, 256, pB2);
  k_pack_b<<<12, 32, 0, stream>>>(W_xp, 128, 36, 48, pB3);
  k_pack_b<<<16, 32, 0, stream>>>(W_out, 128, 64, 64, pB5);

  k_ema<<<1, 384, 0, stream>>>(x, xe);
  k_in_norm<<<ROWS / 8, 256, 0, stream>>>(xe, W_in, b_in, rms_s, h, hn16);
  k_gemm_u<<<ROWS / 32, 256, 0, stream>>>(hn16, pB2, mb_in, u16);
  k_conv<<<ROWS * 128 / 256, 256, 0, stream>>>(u16, convW, convb, xc16);
  k_gemm_xp<<<ROWS / 128, 256, 0, stream>>>(xc16, pB3, xpb);
  k_scan<<<Bn, 128, 0, stream>>>(xpb, xc16, u16, W_dt, b_dt, A_log, Dp, yg16);
  k_gemm_out<<<ROWS / 64, 256, 0, stream>>>(yg16, pB5, b_out, h, hfin);
  k_final<<<ROWS / 256, 256, 0, stream>>>(hfin, W_sim, b_sim, out);
}
